// ResidualVectorQuantizer_76613626626052
// MI455X (gfx1250) — matrix operand order repaired and checked by reading, not compiled, not run on hardware
//
#include <hip/hip_runtime.h>

#define N_TOK      32768
#define DIM        256
#define K_EMB      1024
#define NUM_LAYERS 8
#define TM         32      // rows per block (2 row-tiles of 16)

typedef __attribute__((ext_vector_type(16))) _Float16 v16h;
typedef __attribute__((ext_vector_type(8)))  _Float16 v8h;
typedef __attribute__((ext_vector_type(8)))  float    v8f;

union AFrag { v16h v; v8h h[2]; };

// ---------------------------------------------------------------------------
// Prep: split codebooks into f16 hi/lo, exact f32 ||cb||^2, zero loss accum.
// grid = NUM_LAYERS*K_EMB blocks of 64 threads, one block per codebook entry.
// ---------------------------------------------------------------------------
__global__ __launch_bounds__(64)
void rvq_prep(const float* __restrict__ cb, _Float16* __restrict__ hi,
              _Float16* __restrict__ lo, float* __restrict__ n2,
              float* __restrict__ loss_acc) {
  const int e = blockIdx.x;
  const int t = threadIdx.x;
  __shared__ float red[64];
  const float* src = cb + (size_t)e * DIM;
  float acc = 0.f;
  for (int j = t; j < DIM; j += 64) {
    float v = src[j];
    _Float16 h = (_Float16)v;
    hi[(size_t)e * DIM + j] = h;
    lo[(size_t)e * DIM + j] = (_Float16)(v - (float)h);
    acc += v * v;
  }
  red[t] = acc;
  __syncthreads();
  for (int s = 32; s > 0; s >>= 1) {
    if (t < s) red[t] += red[t + s];
    __syncthreads();
  }
  if (t == 0) n2[e] = red[0];
  if (e == 0 && t == 0) *loss_acc = 0.f;
}

// ---------------------------------------------------------------------------
// One VQ layer: WMMA distances + argmin + fused gather/loss/residual update.
// grid = N_TOK/TM blocks of 256 threads (8 wave32); waves tile 2 (rows) x 4
// (cols); 16 outer iterations cover all K=1024 codebook entries.
// ---------------------------------------------------------------------------
__global__ __launch_bounds__(256)
void rvq_layer(const float* rsrc,                 // residual in (x for layer 0)
               float* res,                        // residual out (workspace)
               const float* __restrict__ cb,      // f32 codebook [K,D] (this layer)
               const _Float16* __restrict__ cbh,  // f16 hi [K,D]
               const _Float16* __restrict__ cbl,  // f16 lo [K,D]
               const float* __restrict__ cbn2,    // [K]
               float* qout,                       // quantized_out accumulator
               float* __restrict__ idx_out,       // indices [N, L] (as float)
               float* __restrict__ loss_acc,
               int layer) {
  __shared__ __attribute__((aligned(32))) _Float16 s_hi[TM * DIM];
  __shared__ __attribute__((aligned(32))) _Float16 s_lo[TM * DIM];
  __shared__ float s_bs[8][16];
  __shared__ int   s_bi[8][16];
  __shared__ int   s_k[TM];
  __shared__ float s_red[256];

  const int tid  = threadIdx.x;
  const int w    = tid >> 5;        // wave 0..7
  const int lane = tid & 31;
  const int l15  = lane & 15;
  const int hf   = lane >> 4;       // half-wave
  const int rt   = w >> 2;          // row-tile 0..1
  const int ct   = w & 3;           // col-tile 0..3
  const size_t rowbase = (size_t)blockIdx.x * TM;
  const float* rs = rsrc + rowbase * DIM;

  // Phase 1: split residual tile into hi/lo f16 in LDS.
  for (int e = tid; e < TM * DIM; e += 256) {
    float v = rs[e];
    _Float16 h = (_Float16)v;
    s_hi[e] = h;
    s_lo[e] = (_Float16)(v - (float)h);
  }
  __syncthreads();

  // Phase 2: A fragments resident in VGPRs (16x32 tile:
  // lane holds 16 contiguous K = hf*16+0..15 per 32-wide K step, matching B).
  AFrag Ah[8], Al[8];
  {
    const int m = rt * 16 + l15;
    const _Float16* ph = s_hi + m * DIM + hf * 16;
    const _Float16* pl = s_lo + m * DIM + hf * 16;
#pragma unroll
    for (int kk = 0; kk < 8; ++kk) {
      Ah[kk].h[0] = *(const v8h*)(ph + kk * 32);
      Ah[kk].h[1] = *(const v8h*)(ph + kk * 32 + 8);
      Al[kk].h[0] = *(const v8h*)(pl + kk * 32);
      Al[kk].h[1] = *(const v8h*)(pl + kk * 32 + 8);
    }
  }

  float bs[8];
  int   bi[8];
#pragma unroll
  for (int r = 0; r < 8; ++r) { bs[r] = 3.4e38f; bi[r] = 0; }

  // Phase 3: distance GEMM. score = ||cb||^2 - 2*dot ( ||f||^2 constant for
  // argmin). Split-precision: Ah*Bh + Ah*Bl + Al*Bh into one f32 accumulator.
  for (int it = 0; it < 16; ++it) {
    const int n = it * 64 + ct * 16 + l15;   // codebook entry for this lane/col
    const _Float16* pbh = cbh + (size_t)n * DIM + hf * 16;
    const _Float16* pbl = cbl + (size_t)n * DIM + hf * 16;
    if (it < 15) {  // prefetch next chunk (global_prefetch_b8)
      __builtin_prefetch(pbh + (size_t)64 * DIM, 0, 1);
      __builtin_prefetch(pbl + (size_t)64 * DIM, 0, 1);
    }
    v8f C = {};
#pragma unroll
    for (int kk = 0; kk < 8; ++kk) {
      // B layout (32x16): lane holds 16 contiguous K at K = hf*16 (32B load)
      v16h Bh = *(const v16h*)(pbh + kk * 32);
      v16h Bl = *(const v16h*)(pbl + kk * 32);
      C = __builtin_amdgcn_wmma_f32_16x16x32_f16(false, Ah[kk].v, false, Bh,
                                                 (short)0, C, false, false);
      C = __builtin_amdgcn_wmma_f32_16x16x32_f16(false, Ah[kk].v, false, Bl,
                                                 (short)0, C, false, false);
      C = __builtin_amdgcn_wmma_f32_16x16x32_f16(false, Al[kk].v, false, Bh,
                                                 (short)0, C, false, false);
    }
    const float c2 = cbn2[n];
#pragma unroll
    for (int r = 0; r < 8; ++r) {
      float s = c2 - 2.0f * C[r];             // row M = r + 8*hf, col n
      if (s < bs[r]) { bs[r] = s; bi[r] = n; } // strict < keeps lowest k
    }
  }

  // Phase 4: argmin across the 16 lanes of each half-wave (same row set).
#pragma unroll
  for (int mask = 1; mask < 16; mask <<= 1) {
#pragma unroll
    for (int r = 0; r < 8; ++r) {
      float os = __shfl_xor(bs[r], mask, 32);
      int   oi = __shfl_xor(bi[r], mask, 32);
      if (os < bs[r] || (os == bs[r] && oi < bi[r])) { bs[r] = os; bi[r] = oi; }
    }
  }
  if (l15 == 0) {
#pragma unroll
    for (int r = 0; r < 8; ++r) {
      s_bs[w][hf * 8 + r] = bs[r];
      s_bi[w][hf * 8 + r] = bi[r];
    }
  }
  __syncthreads();

  // Phase 5: combine the 4 col-tile waves per row; emit index.
  if (tid < TM) {
    const int rrt = tid >> 4, m = tid & 15;
    float best = s_bs[rrt * 4][m];
    int   bk   = s_bi[rrt * 4][m];
    for (int c = 1; c < 4; ++c) {
      float sc = s_bs[rrt * 4 + c][m];
      int   kc = s_bi[rrt * 4 + c][m];
      if (sc < best || (sc == best && kc < bk)) { best = sc; bk = kc; }
    }
    s_k[tid] = bk;
    idx_out[(rowbase + tid) * NUM_LAYERS + layer] = (float)bk;
  }
  __syncthreads();

  // Phase 6: gather codebook row, loss, residual update, quantized accum.
  float lsum = 0.f;
  for (int row = 0; row < TM; ++row) {
    const int k = s_k[row];
    const float q = cb[(size_t)k * DIM + tid];
    const float r = rs[row * DIM + tid];
    const float d = q - r;
    lsum += d * d;
    const size_t g = (rowbase + row) * DIM + tid;
    res[g] = r - q;
    if (layer == 0) qout[g] = q; else qout[g] += q;
  }
  s_red[tid] = lsum;
  __syncthreads();
  for (int s = 128; s > 0; s >>= 1) {
    if (tid < s) s_red[tid] += s_red[tid + s];
    __syncthreads();
  }
  if (tid == 0) atomicAdd(loss_acc, s_red[0]);
}

__global__ void rvq_finalize(const float* __restrict__ loss_acc,
                             float* __restrict__ out_loss) {
  // loss_l = 1.25 * mean((q-r)^2), summed over layers; mean over N*D elements.
  *out_loss = 1.25f * (*loss_acc) / (float)((size_t)N_TOK * DIM);
}

// ---------------------------------------------------------------------------
extern "C" void kernel_launch(void* const* d_in, const int* in_sizes, int n_in,
                              void* d_out, int out_size, void* d_ws,
                              size_t ws_size, hipStream_t stream) {
  const float* x   = (const float*)d_in[0];   // [32,1024,256]
  const float* cbs = (const float*)d_in[1];   // [8,1024,256]

  float* out      = (float*)d_out;
  float* qout     = out;                                   // N*D floats
  float* loss_out = out + (size_t)N_TOK * DIM;             // 1 float
  float* idx_out  = loss_out + 1;                          // N*L floats

  char* ws = (char*)d_ws;
  float* res = (float*)ws;            ws += (size_t)N_TOK * DIM * sizeof(float);
  _Float16* cbh = (_Float16*)ws;      ws += (size_t)NUM_LAYERS * K_EMB * DIM * 2;
  _Float16* cbl = (_Float16*)ws;      ws += (size_t)NUM_LAYERS * K_EMB * DIM * 2;
  float* cbn2 = (float*)ws;           ws += (size_t)NUM_LAYERS * K_EMB * sizeof(float);
  float* loss_acc = (float*)ws;

  rvq_prep<<<NUM_LAYERS * K_EMB, 64, 0, stream>>>(cbs, cbh, cbl, cbn2, loss_acc);

  for (int l = 0; l < NUM_LAYERS; ++l) {
    rvq_layer<<<N_TOK / TM, 256, 0, stream>>>(
        (l == 0) ? x : res, res,
        cbs  + (size_t)l * K_EMB * DIM,
        cbh  + (size_t)l * K_EMB * DIM,
        cbl  + (size_t)l * K_EMB * DIM,
        cbn2 + (size_t)l * K_EMB,
        qout, idx_out, loss_acc, l);
  }

  rvq_finalize<<<1, 1, 0, stream>>>(loss_acc, loss_out);
}
